// MultiHead_9586367005413
// MI455X (gfx1250) — matrix operand order repaired and checked by reading, compiled, not run on hardware
//
#include <hip/hip_runtime.h>
#include <hip/hip_bf16.h>

// ---------------------------------------------------------------------------
// MultiHead attention forward for MI455X (gfx1250, wave32, WMMA).
// All matmul FLOPs go through v_wmma_f32_16x16x32_f16; global->LDS tile
// traffic uses GLOBAL_LOAD_ASYNC_TO_LDS (ASYNCcnt) when the toolchain
// exposes the builtin, with a VGPR-staged fallback otherwise.
// Softmax row-sums are computed on the matrix pipe (P x ones) and the
// exponentials run in exp2 domain so v_exp_f32 is used bare.
// ---------------------------------------------------------------------------

typedef __attribute__((ext_vector_type(16))) _Float16 v16h;
typedef __attribute__((ext_vector_type(8)))  _Float16 v8h;
typedef __attribute__((ext_vector_type(8)))  float    v8f;
typedef int i32x4 __attribute__((vector_size(16)));

#define AS1 __attribute__((address_space(1)))
#define AS3 __attribute__((address_space(3)))

#if defined(__has_builtin)
#if __has_builtin(__builtin_amdgcn_global_load_async_to_lds_b128) && \
    __has_builtin(__builtin_amdgcn_s_wait_asynccnt)
#define HAS_ASYNC_LDS 1
#endif
#endif
#ifndef HAS_ASYNC_LDS
#define HAS_ASYNC_LDS 0
#endif

#define WMMA_F32_F16(a, b, c) \
    __builtin_amdgcn_wmma_f32_16x16x32_f16(false, (a), false, (b), (short)0, (c), false, false)

__device__ __forceinline__ v16h ldfrag(const _Float16* p0, const _Float16* p1) {
    v8h lo = *(const v8h*)p0;
    v8h hi = *(const v8h*)p1;
    return __builtin_shufflevector(lo, hi, 0, 1, 2, 3, 4, 5, 6, 7,
                                           8, 9, 10, 11, 12, 13, 14, 15);
}

__device__ __forceinline__ v8f zero8() {
    v8f z = {0.f, 0.f, 0.f, 0.f, 0.f, 0.f, 0.f, 0.f};
    return z;
}

// 16-byte global -> LDS copy (async DMA when available)
__device__ __forceinline__ void copy16_g2l(const _Float16* g, _Float16* l) {
#if HAS_ASYNC_LDS
    __builtin_amdgcn_global_load_async_to_lds_b128(
        (AS1 i32x4*)(void*)g, (AS3 i32x4*)(void*)l, 0, 0);
#else
    *(uint4*)l = *(const uint4*)g;
#endif
}

__device__ __forceinline__ void wait_async_loads() {
#if HAS_ASYNC_LDS
    __builtin_amdgcn_s_wait_asynccnt(0);
#endif
}

// ---------------------------------------------------------------------------
// fp32 -> f16 elementwise conversion (grid-stride)
// ---------------------------------------------------------------------------
__global__ void cvt_f32_to_f16_kernel(const float* __restrict__ in,
                                      _Float16* __restrict__ out, int n) {
    int i = blockIdx.x * blockDim.x + threadIdx.x;
    int stride = gridDim.x * blockDim.x;
    for (; i < n; i += stride) out[i] = (_Float16)in[i];
}

// ---------------------------------------------------------------------------
// C[m][n] = sum_k A[m][k] * W[n][k] + bias[n]
// A: [M][K] f16 row-major, W: [N][K] f16 row-major ([out,in] torch layout)
// Block: 256 threads = 8 waves; block tile 128x128; K-step 32.
// Double-buffered LDS; async next-tile loads overlap WMMA on current tile.
// ---------------------------------------------------------------------------
template <typename OutT>
__global__ __launch_bounds__(256) void gemm_bias_kernel(
    const _Float16* __restrict__ A, const _Float16* __restrict__ W,
    const float* __restrict__ bias, OutT* __restrict__ C,
    int M, int N, int K) {
    __shared__ _Float16 As[2][128][40];  // padded rows: 80B, 16B-aligned offsets
    __shared__ _Float16 Ws[2][128][40];

    const int tid  = threadIdx.x;
    const int lane = tid & 31;
    const int w    = tid >> 5;
    const int lm   = lane & 15;
    const int hlf  = lane >> 4;
    const int wm   = w >> 2;  // 0..1
    const int wn   = w & 3;   // 0..3
    const int m0   = blockIdx.y * 128;
    const int n0   = blockIdx.x * 128;

    auto load_tiles = [&](int buf, int k0) {
#pragma unroll
        for (int c = 0; c < 2; ++c) {
            int idx = tid + c * 256;  // 0..511
            int r   = idx >> 2;       // 0..127
            int off = (idx & 3) * 8;  // 0,8,16,24 halves
            copy16_g2l(A + (size_t)(m0 + r) * K + k0 + off, &As[buf][r][off]);
            copy16_g2l(W + (size_t)(n0 + r) * K + k0 + off, &Ws[buf][r][off]);
        }
    };

    v8f acc[4][2];
#pragma unroll
    for (int sm = 0; sm < 4; ++sm)
#pragma unroll
        for (int sn = 0; sn < 2; ++sn) acc[sm][sn] = zero8();

    load_tiles(0, 0);

#pragma unroll 2
    for (int k0 = 0; k0 < K; k0 += 32) {
        const int buf = (k0 >> 5) & 1;
        wait_async_loads();   // this wave's tile-(k0) loads complete
        __syncthreads();      // whole block: data visible, prev compute done
        if (k0 + 32 < K) load_tiles(buf ^ 1, k0 + 32);

        v16h af[4], bf[2];
#pragma unroll
        for (int sm = 0; sm < 4; ++sm) {
            const _Float16* p = &As[buf][wm * 64 + sm * 16 + lm][hlf * 8];
            af[sm] = ldfrag(p, p + 16);  // k = hlf*8+i  and  16+hlf*8+i
        }
#pragma unroll
        for (int sn = 0; sn < 2; ++sn) {
            const _Float16* p = &Ws[buf][wn * 32 + sn * 16 + lm][hlf * 8];
            bf[sn] = ldfrag(p, p + 16);  // k = hlf*8+i  and  16+hlf*8+i
        }
#pragma unroll
        for (int sm = 0; sm < 4; ++sm)
#pragma unroll
            for (int sn = 0; sn < 2; ++sn)
                acc[sm][sn] = WMMA_F32_F16(af[sm], bf[sn], acc[sm][sn]);
    }

    float bb[2];
#pragma unroll
    for (int sn = 0; sn < 2; ++sn) bb[sn] = bias[n0 + wn * 32 + sn * 16 + lm];

#pragma unroll
    for (int sm = 0; sm < 4; ++sm)
#pragma unroll
        for (int sn = 0; sn < 2; ++sn)
#pragma unroll
            for (int r = 0; r < 8; ++r) {
                int row = m0 + wm * 64 + sm * 16 + hlf * 8 + r;  // C: m = 8*half + r
                int col = n0 + wn * 32 + sn * 16 + lm;
                C[(size_t)row * N + col] = (OutT)(acc[sm][sn][r] + bb[sn]);
            }
}

// ---------------------------------------------------------------------------
// Flash attention: q,k,v in [B*S][D] layout (head h at column h*64).
// Block = 256 threads = 8 waves; 128 query rows per block; KV tile 64,
// double-buffered with async K loads. Each wave owns 16 query rows x all
// 64 columns -> row max is an intra-wave 16-lane shuffle reduction and the
// row sum rides the matrix pipe (P x ones). exp2-domain online softmax.
// Writes concat layout cat[(b*S+s)*D + h*64 + e] in f16.
// ---------------------------------------------------------------------------
__global__ __launch_bounds__(256) void flash_attn_kernel(
    const _Float16* __restrict__ q, const _Float16* __restrict__ k,
    const _Float16* __restrict__ v, _Float16* __restrict__ cat) {
    constexpr int S = 2048, D = 1024, HD = 64;

    __shared__ _Float16 Qs[128][72];
    __shared__ _Float16 Ks[2][64][72];
    __shared__ _Float16 Vts[2][64][72];  // V transposed: Vts[e][s]
    __shared__ _Float16 Ps[128][72];

    const int tid  = threadIdx.x;
    const int lane = tid & 31;
    const int w    = tid >> 5;       // wave 0..7 -> query rows w*16..w*16+15
    const int lm   = lane & 15;
    const int hlf  = lane >> 4;
    const int bh   = blockIdx.x;     // 0..31
    const int b    = bh >> 4;
    const int h    = bh & 15;
    const int q0   = blockIdx.y * 128;
    const size_t headoff = (size_t)b * S * D + (size_t)h * HD;
    const _Float16* Qh = q + headoff;
    const _Float16* Kh = k + headoff;
    const _Float16* Vh = v + headoff;

    auto load_kv = [&](int buf, int j0) {
#pragma unroll
        for (int c = 0; c < 2; ++c) {
            int idx = tid + c * 256;  // 0..511
            int r   = idx >> 3;       // 0..63
            int off = (idx & 7) * 8;  // 0..56
            copy16_g2l(Kh + (size_t)(j0 + r) * D + off, &Ks[buf][r][off]);
            union { uint4 u; _Float16 hh[8]; } tmp;  // V needs transpose
            tmp.u = *(const uint4*)(Vh + (size_t)(j0 + r) * D + off);
#pragma unroll
            for (int i = 0; i < 8; ++i) Vts[buf][off + i][r] = tmp.hh[i];
        }
    };

    // Load Q tile (128 x 64)
#pragma unroll
    for (int c = 0; c < 4; ++c) {
        int idx = tid + c * 256;   // 0..1023
        int r   = idx >> 3;        // 0..127
        int off = (idx & 7) * 8;   // 0..56
        *(uint4*)(&Qs[r][off]) =
            *(const uint4*)(Qh + (size_t)(q0 + r) * D + off);
    }
    __syncthreads();

    // Preload this wave's Q fragments, pre-scaled by log2(e)/sqrt(HD) so the
    // whole softmax runs in exp2 domain (v_exp_f32 is natively 2^x).
    v16h qa[2];
#pragma unroll
    for (int kk = 0; kk < 2; ++kk) {
        const _Float16* p = &Qs[w * 16 + lm][kk * 32 + hlf * 8];
        qa[kk] = ldfrag(p, p + 16) * (_Float16)0.18033688f;  // 0.125*log2(e)
    }

    v16h ones;
#pragma unroll
    for (int i = 0; i < 16; ++i) ones[i] = (_Float16)1.0f;

    float Mr[8], Lr[8];
#pragma unroll
    for (int r = 0; r < 8; ++r) { Mr[r] = -1e30f; Lr[r] = 0.f; }
    v8f oacc[4];
#pragma unroll
    for (int t = 0; t < 4; ++t) oacc[t] = zero8();

    load_kv(0, 0);

    for (int j0 = 0; j0 < S; j0 += 64) {
        const int buf = (j0 >> 6) & 1;
        wait_async_loads();
        __syncthreads();  // tile visible; prev iteration's reads finished
        if (j0 + 64 < S) load_kv(buf ^ 1, j0 + 64);

        // scores = (Q*scale) * K^T : B[k][n] = Ks[n][k] (contiguous rows)
        v8f sc[4];
#pragma unroll
        for (int t = 0; t < 4; ++t) {
            v8f z = zero8();
#pragma unroll
            for (int kk = 0; kk < 2; ++kk) {
                const _Float16* p = &Ks[buf][t * 16 + lm][kk * 32 + hlf * 8];
                v16h kb = ldfrag(p, p + 16);
                z = WMMA_F32_F16(qa[kk], kb, z);
            }
            sc[t] = z;
        }

        // online softmax (exp2 domain); lane covers rows m = 8*hlf + r
#pragma unroll
        for (int r = 0; r < 8; ++r) {
            float x0 = sc[0][r], x1 = sc[1][r], x2 = sc[2][r], x3 = sc[3][r];
            float mx = fmaxf(fmaxf(x0, x1), fmaxf(x2, x3));
#pragma unroll
            for (int d = 1; d < 16; d <<= 1) mx = fmaxf(mx, __shfl_xor(mx, d, 32));
            float newM  = fmaxf(Mr[r], mx);
            float alpha = exp2f(Mr[r] - newM);
            float p0 = exp2f(x0 - newM), p1 = exp2f(x1 - newM);
            float p2 = exp2f(x2 - newM), p3 = exp2f(x3 - newM);
            Mr[r] = newM;
            Lr[r] *= alpha;  // row sum added below via P x ones on the matrix pipe
#pragma unroll
            for (int t = 0; t < 4; ++t) oacc[t][r] *= alpha;
            int prow = w * 16 + hlf * 8 + r;
            Ps[prow][0 * 16 + lm] = (_Float16)p0;
            Ps[prow][1 * 16 + lm] = (_Float16)p1;
            Ps[prow][2 * 16 + lm] = (_Float16)p2;
            Ps[prow][3 * 16 + lm] = (_Float16)p3;
        }
        __syncthreads();

        // A fragments of P (same-wave rows)
        v16h pa[2];
#pragma unroll
        for (int kk = 0; kk < 2; ++kk) {
            const _Float16* p = &Ps[w * 16 + lm][kk * 32 + hlf * 8];
            pa[kk] = ldfrag(p, p + 16);
        }

        // row sums: P x ones -> every column of the tile holds the row sum
        v8f lsum = zero8();
        lsum = WMMA_F32_F16(pa[0], ones, lsum);
        lsum = WMMA_F32_F16(pa[1], ones, lsum);
#pragma unroll
        for (int r = 0; r < 8; ++r) Lr[r] += lsum[r];

        // O += P * V : B[k][n] = Vts[n][k]
#pragma unroll
        for (int t = 0; t < 4; ++t) {
#pragma unroll
            for (int kk = 0; kk < 2; ++kk) {
                const _Float16* p = &Vts[buf][t * 16 + lm][kk * 32 + hlf * 8];
                v16h vb = ldfrag(p, p + 16);
                oacc[t] = WMMA_F32_F16(pa[kk], vb, oacc[t]);
            }
        }
    }

    // normalize (v_rcp_f32 + muls) and write concat layout
    float inv[8];
#pragma unroll
    for (int r = 0; r < 8; ++r) inv[r] = __builtin_amdgcn_rcpf(Lr[r]);
#pragma unroll
    for (int t = 0; t < 4; ++t)
#pragma unroll
        for (int r = 0; r < 8; ++r) {
            int row = q0 + w * 16 + hlf * 8 + r;
            int col = h * HD + t * 16 + lm;
            float o = oacc[t][r] * inv[r];
            cat[((size_t)b * S + row) * D + col] = (_Float16)o;
        }
}

// ---------------------------------------------------------------------------
// Host launcher
// ---------------------------------------------------------------------------
extern "C" void kernel_launch(void* const* d_in, const int* in_sizes, int n_in,
                              void* d_out, int out_size, void* d_ws, size_t ws_size,
                              hipStream_t stream) {
    (void)in_sizes; (void)n_in; (void)out_size; (void)ws_size;
    const float* V  = (const float*)d_in[0];
    const float* K  = (const float*)d_in[1];
    const float* Q  = (const float*)d_in[2];
    const float* Wv = (const float*)d_in[3];
    const float* bv = (const float*)d_in[4];
    const float* Wk = (const float*)d_in[5];
    const float* bk = (const float*)d_in[6];
    const float* Wq = (const float*)d_in[7];
    const float* bq = (const float*)d_in[8];
    const float* Wo = (const float*)d_in[9];
    const float* bo = (const float*)d_in[10];
    float* out = (float*)d_out;

    const int Bz = 2, S = 2048, Dm = 1024;
    const size_t nBSD = (size_t)Bz * S * Dm;  // 4,194,304
    const size_t nDD  = (size_t)Dm * Dm;      // 1,048,576

    _Float16* ws  = (_Float16*)d_ws;
    _Float16* hV  = ws; ws += nBSD;
    _Float16* hK  = ws; ws += nBSD;
    _Float16* hQ  = ws; ws += nBSD;
    _Float16* hWv = ws; ws += nDD;
    _Float16* hWk = ws; ws += nDD;
    _Float16* hWq = ws; ws += nDD;
    _Float16* hWo = ws; ws += nDD;
    _Float16* qb  = ws; ws += nBSD;
    _Float16* kb  = ws; ws += nBSD;
    _Float16* vb  = ws; ws += nBSD;
    _Float16* cat = hQ;  // hQ dead after q projection; attention runs later

    // fp32 -> f16 conversions
    cvt_f32_to_f16_kernel<<<2048, 256, 0, stream>>>(Q, hQ, (int)nBSD);
    cvt_f32_to_f16_kernel<<<2048, 256, 0, stream>>>(K, hK, (int)nBSD);
    cvt_f32_to_f16_kernel<<<2048, 256, 0, stream>>>(V, hV, (int)nBSD);
    cvt_f32_to_f16_kernel<<<1024, 256, 0, stream>>>(Wq, hWq, (int)nDD);
    cvt_f32_to_f16_kernel<<<1024, 256, 0, stream>>>(Wk, hWk, (int)nDD);
    cvt_f32_to_f16_kernel<<<1024, 256, 0, stream>>>(Wv, hWv, (int)nDD);
    cvt_f32_to_f16_kernel<<<1024, 256, 0, stream>>>(Wo, hWo, (int)nDD);

    // projections: [4096,1024] x [1024,1024]^T
    dim3 gg(1024 / 128, 4096 / 128);  // (8, 32)
    gemm_bias_kernel<_Float16><<<gg, 256, 0, stream>>>(hQ, hWq, bq, qb, 4096, 1024, 1024);
    gemm_bias_kernel<_Float16><<<gg, 256, 0, stream>>>(hK, hWk, bk, kb, 4096, 1024, 1024);
    gemm_bias_kernel<_Float16><<<gg, 256, 0, stream>>>(hV, hWv, bv, vb, 4096, 1024, 1024);

    // attention: grid (B*H, S/128)
    flash_attn_kernel<<<dim3(32, 16), 256, 0, stream>>>(qb, kb, vb, cat);

    // output projection -> fp32
    gemm_bias_kernel<float><<<gg, 256, 0, stream>>>(cat, hWo, bo, out, 4096, 1024, 1024);
}
